// TKANCell_69767448756692
// MI455X (gfx1250) — matrix operand order repaired and checked by reading, not compiled, not run on hardware
//
#include <hip/hip_runtime.h>
#include <hip/hip_bf16.h>

typedef __attribute__((ext_vector_type(16))) _Float16 v16h;
typedef __attribute__((ext_vector_type(8)))  _Float16 v8h;
typedef __attribute__((ext_vector_type(8)))  float    v8f;

#define B_ROWS   4096
#define IN_DIM   256
#define HID      512
#define SUB_IN   256
#define SUB_OUT  256
#define GRID_G   32
#define KAN_K    (2 * SUB_IN * GRID_G)   // 16384
#define GATES_N  (3 * HID)               // 1536

// ---------------- WMMA fragment helpers (wave32, 16x16x32 f16) ----------------

__device__ __forceinline__ v16h load_a_frag(const _Float16* A, int lda) {
  // A-matrix 16x32: lane m = lane%16; K-halves {kb..kb+7} and {kb+16..kb+23}, kb = 8*(lane/16)
  int lane = threadIdx.x & 31;
  int m  = lane & 15;
  int kb = (lane >> 4) << 3;
  const _Float16* p = A + (size_t)m * lda + kb;
  v8h lo = *(const v8h*)(p);
  v8h hi = *(const v8h*)(p + 16);
  return __builtin_shufflevector(lo, hi, 0,1,2,3,4,5,6,7,8,9,10,11,12,13,14,15);
}

__device__ __forceinline__ v16h load_b_frag(const _Float16* Bt, int ldb) {
  // B-matrix 32x16, stored transposed [N][K]: lane n = lane%16; K-halves {kb..kb+7} and {kb+16..kb+23}, kb = 8*(lane/16)
  int lane = threadIdx.x & 31;
  int n  = lane & 15;
  int kb = (lane >> 4) << 3;
  const _Float16* p = Bt + (size_t)n * ldb + kb;
  v8h lo = *(const v8h*)(p);
  v8h hi = *(const v8h*)(p + 16);
  return __builtin_shufflevector(lo, hi, 0,1,2,3,4,5,6,7,8,9,10,11,12,13,14,15);
}

__device__ __forceinline__ v8f wmma_f16(v16h a, v16h b, v8f c) {
  return __builtin_amdgcn_wmma_f32_16x16x32_f16(false, a, false, b, (short)0, c, false, false);
}

// ---------------- weight pack / convert kernels ----------------

__global__ void cvt_f16_kernel(const float* __restrict__ src, _Float16* __restrict__ dst, int n) {
  int i = blockIdx.x * blockDim.x + threadIdx.x;
  if (i < n) dst[i] = (_Float16)src[i];
}

// dst[n*dstK + k0 + k] = src[k*N + n]  (B^T pack)
__global__ void transpose_cvt_kernel(const float* __restrict__ src, _Float16* __restrict__ dst,
                                     int Ksrc, int N, int dstK, int k0) {
  int idx = blockIdx.x * blockDim.x + threadIdx.x;
  if (idx >= Ksrc * N) return;
  int k = idx / N, n = idx - k * N;
  dst[(size_t)n * dstK + k0 + k] = (_Float16)src[(size_t)k * N + n];
}

// fc: [2][SUB_OUT][SUB_IN*GRID] -> Wfc_t[o][t*8192 + j]
__global__ void fc_pack_kernel(const float* __restrict__ fc, _Float16* __restrict__ dst) {
  int idx = blockIdx.x * blockDim.x + threadIdx.x;   // 2*256*8192
  if (idx >= 2 * SUB_OUT * SUB_IN * GRID_G) return;
  int t   = idx >> 21;
  int rem = idx & ((1 << 21) - 1);
  int o   = rem >> 13;
  int j   = rem & 8191;
  dst[(size_t)o * KAN_K + (size_t)t * 8192 + j] = (_Float16)fc[idx];
}

// ---------------- generic WMMA GEMM, all dims compile-time ----------------
// Wave computes 64x32 output tile (4 M-frags x 2 N-frags).
// A split at KSPLIT: k < KSPLIT -> A0 (stride LDA0), else A1 (stride LDA1).
// Bt is [N][K] f16. MODE: 0 sigmoid(acc+bias); 1 plain; 2 sigmoid(acc+bias)*extra.

template<int LDA0, int KSPLIT, int LDA1, int N, int K, int MODE>
__global__ void wmma_gemm_kernel(const _Float16* __restrict__ A0,
                                 const _Float16* __restrict__ A1,
                                 const _Float16* __restrict__ Bt,
                                 const float* __restrict__ bias,
                                 const float* __restrict__ extra,
                                 float* __restrict__ out) {
  constexpr int MT = 64;
  constexpr int NT = N >> 5;
  int wave = blockIdx.x * (blockDim.x >> 5) + (threadIdx.x >> 5);
  int tm = wave / NT, tn = wave - tm * NT;
  int m0 = tm * MT, n0 = tn << 5;

  v8f zero = {0.f,0.f,0.f,0.f,0.f,0.f,0.f,0.f};
  v8f acc[4][2];
  #pragma unroll
  for (int i = 0; i < 4; ++i) { acc[i][0] = zero; acc[i][1] = zero; }

  const _Float16* bt0 = Bt + (size_t)n0 * K;
  const _Float16* bt1 = Bt + (size_t)(n0 + 16) * K;
  const _Float16* ab  = A0 + (size_t)m0 * LDA0;

  #pragma unroll 2
  for (int k = 0; k < KSPLIT; k += 32) {
    v16h b0 = load_b_frag(bt0 + k, K);
    v16h b1 = load_b_frag(bt1 + k, K);
    #pragma unroll
    for (int i = 0; i < 4; ++i) {
      v16h a = load_a_frag(ab + (size_t)(16 * i) * LDA0 + k, LDA0);
      acc[i][0] = wmma_f16(a, b0, acc[i][0]);
      acc[i][1] = wmma_f16(a, b1, acc[i][1]);
    }
  }
  if constexpr (KSPLIT < K) {
    const _Float16* ab1 = A1 + (size_t)m0 * LDA1;
    #pragma unroll 2
    for (int k = KSPLIT; k < K; k += 32) {
      int kk = k - KSPLIT;
      v16h b0 = load_b_frag(bt0 + k, K);
      v16h b1 = load_b_frag(bt1 + k, K);
      #pragma unroll
      for (int i = 0; i < 4; ++i) {
        v16h a = load_a_frag(ab1 + (size_t)(16 * i) * LDA1 + kk, LDA1);
        acc[i][0] = wmma_f16(a, b0, acc[i][0]);
        acc[i][1] = wmma_f16(a, b1, acc[i][1]);
      }
    }
  }

  int lane = threadIdx.x & 31;
  int n  = lane & 15;
  int mb = (lane >> 4) << 3;
  #pragma unroll
  for (int i = 0; i < 4; ++i) {
    #pragma unroll
    for (int j = 0; j < 2; ++j) {
      #pragma unroll
      for (int r = 0; r < 8; ++r) {
        int row = m0 + i * 16 + mb + r;
        int col = n0 + j * 16 + n;
        float v = acc[i][j][r];
        size_t oidx = (size_t)row * N + col;
        if constexpr (MODE == 0) {
          v += bias[col];
          out[oidx] = 1.0f / (1.0f + __expf(-v));
        } else if constexpr (MODE == 1) {
          out[oidx] = v;
        } else {
          v += bias[col];
          float o = 1.0f / (1.0f + __expf(-v));
          out[oidx] = o * extra[oidx];
        }
      }
    }
  }
}

// ---------------- Fourier-KAN GEMM: double-buffered LDS cos/sin features ----------------
// Block: 256 threads (8 waves), 32 rows x all 256 out columns. K = 16384.
// z rows staged in LDS once; features: branchless sin(ang + phase), vector b128 LDS stores.

#define KAN_ROWS 32
#define FEAT_CH  64
#define FEAT_LD  72   // 144B stride = 9 * 16B -> aligned b128 LDS loads

__global__ void kan_gemm_kernel(const float* __restrict__ agg_in,
                                const _Float16* __restrict__ Wfc,
                                const float* __restrict__ fb,
                                const float* __restrict__ rk,      // [2*SUB_OUT]: rk_h | rk_x
                                const float* __restrict__ s0,
                                _Float16* __restrict__ subout16,
                                float* __restrict__ out_s) {
  __shared__ __align__(16) _Float16 feat[2][KAN_ROWS * FEAT_LD];
  __shared__ __align__(16) float zbuf[KAN_ROWS * SUB_IN];          // 32 KB
  int row0 = blockIdx.x * KAN_ROWS;
  int tid  = threadIdx.x;
  int wv   = tid >> 5;
  int lane = tid & 31;
  int n0   = wv << 5;            // this wave's 32 output columns

  // ---- stage the block's agg_in rows into LDS (coalesced float4) ----
  {
    const float4* src = (const float4*)(agg_in + (size_t)row0 * SUB_IN);
    float4* dst = (float4*)zbuf;
    #pragma unroll
    for (int q = 0; q < (KAN_ROWS * SUB_IN / 4) / 256; ++q)
      dst[tid + 256 * q] = src[tid + 256 * q];
  }

  v8f zero = {0.f,0.f,0.f,0.f,0.f,0.f,0.f,0.f};
  v8f a00 = zero, a01 = zero, a10 = zero, a11 = zero;

  // feature-gen mapping: thread -> row fr = tid/8, 8 contiguous k within one 32-block
  int fr  = tid >> 3;
  int kk0 = (tid & 7) << 3;
  int g0  = kk0 & 31;
  const float* zrow = zbuf + fr * SUB_IN + (kk0 >> 5);

  auto gen = [&](int kc, int buf) {
    // cos(x) = sin(x + pi/2): uniform phase select per chunk, no per-element branch
    float phase = (kc < 8192) ? 1.5707963267948966f : 0.0f;
    int ii = (kc & 8191) >> 5;
    float z = zrow[ii];
    v8h vals;
    #pragma unroll
    for (int j = 0; j < 8; ++j) {
      float ang = fmaf(z, (float)(g0 + j + 1), phase);
      vals[j] = (_Float16)__sinf(ang);
    }
    *(v8h*)&feat[buf][fr * FEAT_LD + kk0] = vals;   // one ds_store_b128
  };

  const _Float16* wb0 = Wfc + (size_t)n0 * KAN_K;
  const _Float16* wb1 = Wfc + (size_t)(n0 + 16) * KAN_K;

  __syncthreads();               // zbuf ready
  gen(0, 0);
  __syncthreads();

  int buf = 0;
  for (int kc = 0; kc < KAN_K; kc += FEAT_CH) {
    // stage next chunk's features into the spare buffer (uniform branch, EXEC stays full)
    if (kc + FEAT_CH < KAN_K) gen(kc + FEAT_CH, buf ^ 1);
    // prefetch next chunk of B into near caches (global_prefetch_b8)
    __builtin_prefetch(wb0 + kc + FEAT_CH, 0, 1);
    __builtin_prefetch(wb1 + kc + FEAT_CH, 0, 1);
    #pragma unroll
    for (int ks = 0; ks < FEAT_CH; ks += 32) {
      v16h fa0 = load_a_frag(&feat[buf][0 * FEAT_LD + ks], FEAT_LD);
      v16h fa1 = load_a_frag(&feat[buf][16 * FEAT_LD + ks], FEAT_LD);
      v16h fb0 = load_b_frag(wb0 + kc + ks, KAN_K);
      v16h fb1 = load_b_frag(wb1 + kc + ks, KAN_K);
      a00 = wmma_f16(fa0, fb0, a00);
      a01 = wmma_f16(fa0, fb1, a01);
      a10 = wmma_f16(fa1, fb0, a10);
      a11 = wmma_f16(fa1, fb1, a11);
    }
    __syncthreads();
    buf ^= 1;
  }

  int n  = lane & 15;
  int mb = (lane >> 4) << 3;
  v8f accs[4] = {a00, a01, a10, a11};
  #pragma unroll
  for (int f = 0; f < 4; ++f) {
    int i = f >> 1, j = f & 1;
    #pragma unroll
    for (int r = 0; r < 8; ++r) {
      int row = row0 + i * 16 + mb + r;
      int col = n0 + j * 16 + n;
      float v = accs[f][r] + fb[col];
      size_t idx = (size_t)row * SUB_OUT + col;
      subout16[idx] = (_Float16)v;                         // f16 A-input for out-gate GEMM
      out_s[idx] = rk[col] * v + rk[SUB_OUT + col] * s0[idx];
    }
  }
}

// ---------------- elementwise: cell state ----------------

__global__ void cell_kernel(const float* __restrict__ gates, const float* __restrict__ c0,
                            float* __restrict__ out_c, float* __restrict__ tanhc) {
  int idx = blockIdx.x * blockDim.x + threadIdx.x;
  if (idx >= B_ROWS * HID) return;
  int row = idx >> 9;          // / 512
  int col = idx & (HID - 1);
  size_t base = (size_t)row * GATES_N;
  float ig = gates[base + col];
  float fg = gates[base + HID + col];
  float cc = gates[base + 2 * HID + col];
  float c  = fg * c0[idx] + ig * tanhf(cc);
  out_c[idx] = c;
  tanhc[idx] = tanhf(c);
}

// ---------------- layer norm: one wave32 per row of 512 ----------------

__global__ void ln_kernel(const float* __restrict__ hpre,
                          const float* __restrict__ g, const float* __restrict__ b,
                          float* __restrict__ out_h) {
  int wv   = threadIdx.x >> 5;
  int lane = threadIdx.x & 31;
  int row  = blockIdx.x * 8 + wv;
  const float* p = hpre + (size_t)row * HID;
  float v[16];
  float s = 0.f, ss = 0.f;
  #pragma unroll
  for (int j = 0; j < 16; ++j) {
    v[j] = p[j * 32 + lane];
    s  += v[j];
    ss += v[j] * v[j];
  }
  #pragma unroll
  for (int off = 16; off >= 1; off >>= 1) {
    s  += __shfl_xor(s, off, 32);
    ss += __shfl_xor(ss, off, 32);
  }
  float mu  = s * (1.f / (float)HID);
  float var = ss * (1.f / (float)HID) - mu * mu;
  float inv = rsqrtf(var + 1e-5f);
  float* q = out_h + (size_t)row * HID;
  #pragma unroll
  for (int j = 0; j < 16; ++j) {
    int col = j * 32 + lane;
    q[col] = (v[j] - mu) * inv * g[col] + b[col];
  }
}

// ---------------- host launch ----------------

extern "C" void kernel_launch(void* const* d_in, const int* in_sizes, int n_in,
                              void* d_out, int out_size, void* d_ws, size_t ws_size,
                              hipStream_t stream) {
  const float* x       = (const float*)d_in[0];
  const float* h0      = (const float*)d_in[1];
  const float* c0      = (const float*)d_in[2];
  const float* s0      = (const float*)d_in[3];
  const float* kernel_ = (const float*)d_in[4];
  const float* rkernel = (const float*)d_in[5];
  const float* bias    = (const float*)d_in[6];
  const float* sub_kx  = (const float*)d_in[7];
  const float* sub_kh  = (const float*)d_in[8];
  const float* tkan_k  = (const float*)d_in[9];
  const float* fc      = (const float*)d_in[10];
  const float* fb      = (const float*)d_in[11];
  const float* agg_w   = (const float*)d_in[12];
  const float* agg_b   = (const float*)d_in[13];
  const float* ln_g    = (const float*)d_in[14];
  const float* ln_b    = (const float*)d_in[15];

  float* out_h = (float*)d_out;
  float* out_c = out_h + (size_t)B_ROWS * HID;
  float* out_s = out_c + (size_t)B_ROWS * HID;

  // workspace carve (256B aligned)
  char* w = (char*)d_ws;
  auto alloc = [&](size_t bytes) { void* p = (void*)w; w += (bytes + 255) & ~(size_t)255; return p; };
  _Float16* x16      = (_Float16*)alloc((size_t)B_ROWS * IN_DIM * 2);
  _Float16* h016     = (_Float16*)alloc((size_t)B_ROWS * HID * 2);
  _Float16* s016     = (_Float16*)alloc((size_t)B_ROWS * SUB_OUT * 2);
  _Float16* Wg       = (_Float16*)alloc((size_t)GATES_N * (IN_DIM + HID) * 2);
  _Float16* Wsub     = (_Float16*)alloc((size_t)SUB_IN * (IN_DIM + SUB_OUT) * 2);
  _Float16* Wfc      = (_Float16*)alloc((size_t)SUB_OUT * KAN_K * 2);
  _Float16* Wagg     = (_Float16*)alloc((size_t)HID * SUB_OUT * 2);
  float*    gatesig  = (float*)alloc((size_t)B_ROWS * GATES_N * 4);
  float*    tanhc    = (float*)alloc((size_t)B_ROWS * HID * 4);
  float*    agg_in   = (float*)alloc((size_t)B_ROWS * SUB_IN * 4);
  _Float16* subout16 = (_Float16*)alloc((size_t)B_ROWS * SUB_OUT * 2);
  float*    hpre     = (float*)alloc((size_t)B_ROWS * HID * 4);
  (void)ws_size; (void)n_in; (void)in_sizes; (void)out_size;

  const int T = 256;
  auto cdiv = [](int a, int b) { return (a + b - 1) / b; };

  // ---- pack / convert ----
  cvt_f16_kernel<<<cdiv(B_ROWS * IN_DIM, T), T, 0, stream>>>(x, x16, B_ROWS * IN_DIM);
  cvt_f16_kernel<<<cdiv(B_ROWS * HID, T), T, 0, stream>>>(h0, h016, B_ROWS * HID);
  cvt_f16_kernel<<<cdiv(B_ROWS * SUB_OUT, T), T, 0, stream>>>(s0, s016, B_ROWS * SUB_OUT);
  transpose_cvt_kernel<<<cdiv(IN_DIM * GATES_N, T), T, 0, stream>>>(kernel_, Wg, IN_DIM, GATES_N, IN_DIM + HID, 0);
  transpose_cvt_kernel<<<cdiv(HID * GATES_N, T), T, 0, stream>>>(rkernel, Wg, HID, GATES_N, IN_DIM + HID, IN_DIM);
  transpose_cvt_kernel<<<cdiv(IN_DIM * SUB_IN, T), T, 0, stream>>>(sub_kx, Wsub, IN_DIM, SUB_IN, IN_DIM + SUB_OUT, 0);
  transpose_cvt_kernel<<<cdiv(SUB_OUT * SUB_IN, T), T, 0, stream>>>(sub_kh, Wsub, SUB_OUT, SUB_IN, IN_DIM + SUB_OUT, IN_DIM);
  fc_pack_kernel<<<cdiv(2 * SUB_OUT * SUB_IN * GRID_G, T), T, 0, stream>>>(fc, Wfc);
  transpose_cvt_kernel<<<cdiv(SUB_OUT * HID, T), T, 0, stream>>>(agg_w, Wagg, SUB_OUT, HID, SUB_OUT, 0);

  // ---- gates GEMM: [4096 x 768] x [768 x 1536], sigmoid epilogue ----
  {
    int waves = (B_ROWS / 64) * (GATES_N / 32);          // 3072
    wmma_gemm_kernel<IN_DIM, IN_DIM, HID, GATES_N, IN_DIM + HID, 0>
        <<<waves / 8, T, 0, stream>>>(x16, h016, Wg, bias, nullptr, gatesig);
  }
  // ---- cell state ----
  cell_kernel<<<cdiv(B_ROWS * HID, T), T, 0, stream>>>(gatesig, c0, out_c, tanhc);

  // ---- sub-mix GEMM: [4096 x 512] x [512 x 256] -> agg_in ----
  {
    int waves = (B_ROWS / 64) * (SUB_IN / 32);           // 512
    wmma_gemm_kernel<IN_DIM, IN_DIM, SUB_OUT, SUB_IN, IN_DIM + SUB_OUT, 1>
        <<<waves / 8, T, 0, stream>>>(x16, s016, Wsub, nullptr, nullptr, agg_in);
  }

  // ---- Fourier-KAN GEMM (double-buffered LDS cos/sin features) + new_s ----
  kan_gemm_kernel<<<B_ROWS / KAN_ROWS, T, 0, stream>>>(
      agg_in, Wfc, fb, tkan_k, s0, subout16, out_s);

  // ---- out-gate GEMM: [4096 x 256] x [256 x 512], sigmoid * tanh(c) ----
  {
    int waves = (B_ROWS / 64) * (HID / 32);              // 1024
    wmma_gemm_kernel<SUB_OUT, SUB_OUT, SUB_OUT, HID, SUB_OUT, 2>
        <<<waves / 8, T, 0, stream>>>(subout16, subout16, Wagg, agg_b, tanhc, hpre);
  }

  // ---- layer norm ----
  ln_kernel<<<B_ROWS / 8, T, 0, stream>>>(hpre, ln_g, ln_b, out_h);
}
